// FullyContextualEmbedding_11751030522507
// MI455X (gfx1250) — matrix operand order repaired and checked by reading, not compiled, not run on hardware
//
#include <hip/hip_runtime.h>
#include <hip/hip_bf16.h>
#include <math.h>

typedef __attribute__((ext_vector_type(16))) __bf16 v16bf;
typedef __attribute__((ext_vector_type(8)))  float  v8f;

#define NROWS 1024
#define DDIM  512
#define KCTX  64
#define KDIM  1536   // 3*D : [f | r | h]
#define GDIM  2048   // 4*D gates

__device__ __forceinline__ unsigned short f32_to_bf16(float f) {
  union { float f; unsigned int u; } v; v.f = f;
  unsigned int u = v.u;
  unsigned int r = (u + 0x7FFFu + ((u >> 16) & 1u)) >> 16;  // RNE
  return (unsigned short)r;
}

__device__ __forceinline__ float sigmoidf_(float x) {
  return 1.0f / (1.0f + __expf(-x));
}

// h0 = f, c0 = 0, x[:,0:512]=bf16(f), x[:,1024:1536]=bf16(h0), bsum=b_ih+b_hh
__global__ void prep_state(const float* __restrict__ f,
                           float* __restrict__ h, float* __restrict__ c,
                           unsigned short* __restrict__ x,
                           const float* __restrict__ b_ih,
                           const float* __restrict__ b_hh,
                           float* __restrict__ bsum) {
  int i = blockIdx.x * blockDim.x + threadIdx.x;
  if (i < NROWS * DDIM) {
    float v = f[i];
    int n = i >> 9, j = i & 511;
    h[i] = v;
    c[i] = 0.0f;
    unsigned short bv = f32_to_bf16(v);
    x[n * KDIM + j]        = bv;   // f columns (static across steps)
    x[n * KDIM + 1024 + j] = bv;   // h columns (h0 = f)
  }
  if (i < GDIM) bsum[i] = b_ih[i] + b_hh[i];
}

// WT[n][k] = bf16( k<1024 ? W_ih[n][k] : W_hh[n][k-1024] )   [2048 x 1536]
__global__ void prep_weights(const float* __restrict__ W_ih,
                             const float* __restrict__ W_hh,
                             unsigned short* __restrict__ WT) {
  int i = blockIdx.x * blockDim.x + threadIdx.x;
  if (i >= GDIM * KDIM) return;
  int n = i / KDIM, k = i % KDIM;
  float v = (k < 1024) ? W_ih[n * 1024 + k] : W_hh[n * 512 + (k - 1024)];
  WT[i] = f32_to_bf16(v);
}

// per-row softmax(h@G^T) @ G -> bf16 r into x[:,512:1024]; f32 VALU
__global__ __launch_bounds__(64)
void attn_kernel(const float* __restrict__ h, const float* __restrict__ G,
                 unsigned short* __restrict__ x) {
  int n = blockIdx.x;
  int t = threadIdx.x;              // 0..63 == context index
  __shared__ float hs[DDIM];
  __shared__ float logits[KCTX];
  __shared__ float red[KCTX];

  for (int j = t; j < DDIM; j += 64) hs[j] = h[n * DDIM + j];
  __syncthreads();

  const float* g = G + t * DDIM;
  float acc = 0.0f;
  #pragma unroll 4
  for (int j = 0; j < DDIM; j += 4) {
    acc += hs[j] * g[j] + hs[j + 1] * g[j + 1] +
           hs[j + 2] * g[j + 2] + hs[j + 3] * g[j + 3];
  }
  logits[t] = acc;
  __syncthreads();

  float m = -3.0e38f;
  for (int k = 0; k < KCTX; ++k) m = fmaxf(m, logits[k]);
  float e = __expf(acc - m);
  red[t] = e;
  __syncthreads();
  float s = 0.0f;
  for (int k = 0; k < KCTX; ++k) s += red[k];
  __syncthreads();
  logits[t] = e / s;                // a_t
  __syncthreads();

  float r[8] = {0.f,0.f,0.f,0.f,0.f,0.f,0.f,0.f};
  for (int k = 0; k < KCTX; ++k) {
    float ak = logits[k];
    const float* gk = G + k * DDIM + t * 8;
    #pragma unroll
    for (int q = 0; q < 8; ++q) r[q] += ak * gk[q];
  }
  unsigned short* xr = x + n * KDIM + DDIM + t * 8;
  #pragma unroll
  for (int q = 0; q < 8; ++q) xr[q] = f32_to_bf16(r[q]);
}

// gates[1024x2048] = x @ W_cat + bsum via v_wmma_f32_16x16x32_bf16
// Double-buffered LDS filled by GLOBAL_LOAD_ASYNC_TO_LDS_B128 (ASYNCcnt).
#define BM 128
#define BN 128
#define BK 32
#define LDSS 40            // padded bf16 row stride (80B, 16B aligned)
#define NIT (KDIM / BK)    // 48

union FragU { uint4 q[2]; v16bf v; };

// Async DMA of 16B from global to LDS; per-lane addresses; tracked by ASYNCcnt.
__device__ __forceinline__ void async_copy16(const unsigned short* gp,
                                             const unsigned short* lp) {
  unsigned int lds_off = (unsigned int)(unsigned long long)(const void*)lp;
  unsigned long long gaddr = (unsigned long long)(const void*)gp;
  asm volatile("global_load_async_to_lds_b128 %0, %1, off"
               :: "v"(lds_off), "v"(gaddr) : "memory");
}

__device__ __forceinline__ void wait_asynccnt0() {
  asm volatile("s_wait_asynccnt 0x0" ::: "memory");
}

__global__ __launch_bounds__(256)
void gates_gemm(const unsigned short* __restrict__ X,   // [1024][1536]
                const unsigned short* __restrict__ WT,  // [2048][1536]
                const float* __restrict__ bsum,         // [2048]
                float* __restrict__ gates) {            // [1024][2048]
  __shared__ unsigned short As[2][BM * LDSS];
  __shared__ unsigned short Bs[2][BN * LDSS];

  const int tid   = threadIdx.x;
  const int lane  = tid & 31;
  const int wave  = tid >> 5;       // 0..7
  const int waveM = wave >> 2;      // 0..1
  const int waveN = wave & 3;       // 0..3
  const int blockM = blockIdx.y * BM;
  const int blockN = blockIdx.x * BN;

  v8f acc[4][2] = {};

  const int lrow16 = lane & 15;
  const int hiHalf = lane >> 4;             // 0 | 1
  const int aOff   = hiHalf * 8;            // bf16 offset {0,8}  (A K split)
  const int bOff   = hiHalf * 8;            // bf16 offset {0,8}  (B K split)

  // Per-thread staging coords: 512 x 16B chunks per tile, 2 per thread.
  const int chunkBase = tid * 2;

  // Issue async fills of A/B tiles for K-iteration `it` into buffer it&1.
  auto issue_tile = [&](int it) {
    const int buf = it & 1;
    const int kk  = it * BK;
    #pragma unroll
    for (int cI = 0; cI < 2; ++cI) {
      int chunk = chunkBase + cI;
      int row = chunk >> 2;
      int col = (chunk & 3) * 8;
      async_copy16(X  + (blockM + row) * KDIM + kk + col,
                   &As[buf][row * LDSS + col]);
      async_copy16(WT + (blockN + row) * KDIM + kk + col,
                   &Bs[buf][row * LDSS + col]);
    }
  };

  issue_tile(0);

  for (int it = 0; it < NIT; ++it) {
    wait_asynccnt0();          // my fills of buffer `it` landed in LDS
    __syncthreads();           // everyone's fills landed; buf[it-1] reads done
    if (it + 1 < NIT) issue_tile(it + 1);   // overlaps WMMA below

    const int buf = it & 1;
    const unsigned short* Ab = &As[buf][0];
    const unsigned short* Bb = &Bs[buf][0];

    FragU afr[4], bfr[2];
    #pragma unroll
    for (int tm = 0; tm < 4; ++tm) {
      const unsigned short* base = Ab + (waveM * 64 + tm * 16 + lrow16) * LDSS;
      afr[tm].q[0] = *(const uint4*)(base + aOff);
      afr[tm].q[1] = *(const uint4*)(base + 16 + aOff);
    }
    #pragma unroll
    for (int tn = 0; tn < 2; ++tn) {
      const unsigned short* base = Bb + (waveN * 32 + tn * 16 + lrow16) * LDSS;
      bfr[tn].q[0] = *(const uint4*)(base + bOff);
      bfr[tn].q[1] = *(const uint4*)(base + 16 + bOff);
    }
    #pragma unroll
    for (int tm = 0; tm < 4; ++tm)
      #pragma unroll
      for (int tn = 0; tn < 2; ++tn)
        acc[tm][tn] = __builtin_amdgcn_wmma_f32_16x16x32_bf16(
            false, afr[tm].v, false, bfr[tn].v,
            (short)0, acc[tm][tn], false, false);
  }

  const int rowBase = blockM + waveM * 64;
  const int colBase = blockN + waveN * 32;
  const int rOff = hiHalf * 8;              // VGPR i -> M=i | M=8+i
  #pragma unroll
  for (int tm = 0; tm < 4; ++tm) {
    #pragma unroll
    for (int tn = 0; tn < 2; ++tn) {
      int col = colBase + tn * 16 + lrow16;
      float bias = bsum[col];
      #pragma unroll
      for (int i = 0; i < 8; ++i) {
        int row = rowBase + tm * 16 + i + rOff;
        gates[row * GDIM + col] = acc[tm][tn][i] + bias;
      }
    }
  }
}

// LSTM cell elementwise + residual; refresh h f32 and bf16 h-columns of x
__global__ void cell_kernel(const float* __restrict__ gates,
                            const float* __restrict__ f,
                            float* __restrict__ h, float* __restrict__ c,
                            unsigned short* __restrict__ x) {
  int i = blockIdx.x * blockDim.x + threadIdx.x;
  if (i >= NROWS * DDIM) return;
  int n = i >> 9, j = i & 511;
  const float* gr = gates + n * GDIM;
  float ig = sigmoidf_(gr[j]);
  float fg = sigmoidf_(gr[512 + j]);
  float gg = tanhf(gr[1024 + j]);
  float og = sigmoidf_(gr[1536 + j]);
  float cn = fg * c[i] + ig * gg;
  float hn = og * tanhf(cn) + f[i];   // residual
  c[i] = cn;
  h[i] = hn;
  x[n * KDIM + 1024 + j] = f32_to_bf16(hn);
}

extern "C" void kernel_launch(void* const* d_in, const int* in_sizes, int n_in,
                              void* d_out, int out_size, void* d_ws, size_t ws_size,
                              hipStream_t stream) {
  const float* f    = (const float*)d_in[0];
  const float* G    = (const float*)d_in[1];
  const float* W_ih = (const float*)d_in[2];
  const float* W_hh = (const float*)d_in[3];
  const float* b_ih = (const float*)d_in[4];
  const float* b_hh = (const float*)d_in[5];

  char* ws = (char*)d_ws;
  size_t off = 0;
  unsigned short* X  = (unsigned short*)(ws + off); off += (size_t)NROWS * KDIM * 2;
  unsigned short* WT = (unsigned short*)(ws + off); off += (size_t)GDIM  * KDIM * 2;
  float* h     = (float*)(ws + off); off += (size_t)NROWS * DDIM * 4;
  float* c     = (float*)(ws + off); off += (size_t)NROWS * DDIM * 4;
  float* gates = (float*)(ws + off); off += (size_t)NROWS * GDIM * 4;
  float* bsum  = (float*)(ws + off); off += (size_t)GDIM * 4;

  prep_state<<<(NROWS * DDIM + 255) / 256, 256, 0, stream>>>(f, h, c, X, b_ih, b_hh, bsum);
  prep_weights<<<(GDIM * KDIM + 255) / 256, 256, 0, stream>>>(W_ih, W_hh, WT);

  for (int step = 0; step < KCTX; ++step) {
    attn_kernel<<<NROWS, 64, 0, stream>>>(h, G, X);
    gates_gemm<<<dim3(GDIM / BN, NROWS / BM), 256, 0, stream>>>(X, WT, bsum, gates);
    cell_kernel<<<(NROWS * DDIM + 255) / 256, 256, 0, stream>>>(gates, f, h, c, X);
  }

  hipMemcpyAsync(d_out, h, (size_t)NROWS * DDIM * sizeof(float),
                 hipMemcpyDeviceToDevice, stream);
}
